// VectorQuantizer_67851893342891
// MI455X (gfx1250) — matrix operand order repaired and checked by reading, not compiled, not run on hardware
//
#include <hip/hip_runtime.h>

// ---------- problem constants ----------
#define BATCH    4
#define CCH      256        // embedding dim
#define DHW      8192       // 256 blocks x 32 rows per b
#define NROWS    65536      // 8*DHW (b runs 0..7)
#define KCODES   4096
#define NELEM    16777216   // NROWS*CCH
#define TILE_R   128        // z-rows per block (N dimension of WMMA)
#define TILE_C   128        // codes per pass   (M dimension of WMMA)
#define NPASS    (KCODES / TILE_C)   // 32
#define Z_PITCH  136        // halves (272B = 17*16: aligned, 4-bank lane stride)
#define W_PITCH  264        // halves (528B = 33*16: aligned, 4-bank lane stride)
#define SMEM_HALVES (CCH * Z_PITCH + 2 * TILE_C * W_PITCH)
#define SMEM_BYTES  (SMEM_HALVES * 2)          // 204800 B (<320KB WGP LDS)

typedef __attribute__((ext_vector_type(16))) _Float16 v16h;
typedef __attribute__((ext_vector_type(8)))  _Float16 v8h;
typedef __attribute__((ext_vector_type(8)))  float    v8f;

// ---- gfx1250 async global->LDS path (guarded; falls back to uint4 copies) ----
#if defined(__has_builtin)
#if __has_builtin(__builtin_amdgcn_global_load_async_to_lds_b128) && \
    __has_builtin(__builtin_amdgcn_s_wait_asynccnt)
#define USE_ASYNC 1
#endif
#if __has_builtin(__builtin_amdgcn_tensor_load_to_lds) && \
    __has_builtin(__builtin_amdgcn_s_wait_tensorcnt)
#define USE_TDM 1
#endif
#endif
#ifndef USE_ASYNC
#define USE_ASYNC 0
#endif
#ifndef USE_TDM
#define USE_TDM 0
#endif

// per the compiler diagnostic: param0 is `int __vector(4) __device__ *`
typedef int vb128 __attribute__((vector_size(16)));
typedef __attribute__((address_space(1))) vb128* gvp;   // global (AS1)
typedef __attribute__((address_space(3))) vb128* lvp;   // LDS (AS3)

__device__ __forceinline__ void cp16(const _Float16* gsrc, _Float16* ldst) {
#if USE_ASYNC
  __builtin_amdgcn_global_load_async_to_lds_b128((gvp)gsrc, (lvp)ldst, 0, 0);
#else
  *(uint4*)ldst = *(const uint4*)gsrc;
#endif
}

#if USE_TDM
typedef unsigned tv4u __attribute__((ext_vector_type(4)));
typedef int      tv8i __attribute__((ext_vector_type(8)));
typedef int      tv4i __attribute__((ext_vector_type(4)));

// One TDM descriptor moves a whole 64KB W tile (16384 dwords) global->LDS,
// with hardware padding: 4 dwords pad per 128 dwords (row) => W_PITCH=264h.
__device__ __forceinline__ void tdm_load_tile(const _Float16* gsrc,
                                              _Float16* ldst) {
  unsigned long long ga = (unsigned long long)(uintptr_t)gsrc;
  unsigned lds = (unsigned)(uintptr_t)ldst;      // low 32 bits = LDS offset
  tv4u g0;
  g0.x = 1u;                                     // count=1, user D#
  g0.y = lds;                                    // lds_addr (bytes)
  g0.z = (unsigned)(ga & 0xFFFFFFFFu);           // global_addr[31:0]
  g0.w = (unsigned)((ga >> 32) & 0x01FFFFFFu) | 0x80000000u; // [56:32]|type=2
  tv8i g1;
  g1[0] = (2 << 16)      // data_size = 4B
        | (1 << 20)      // pad_enable
        | (6 << 22)      // pad_interval: every 128 dwords (one 256-half row)
        | (3 << 25);     // pad_amount: 4 dwords (8 halves)
  g1[1] = (int)(16384u << 16);  // tensor_dim0[15:0]=16384 (in hi half of word)
  g1[2] = 0;                    // tensor_dim0[31:16]=0, tensor_dim1 lo=0
  g1[3] = (int)(16384u << 16);  // tile_dim0 = 16384 dwords
  g1[4] = 0;                    // tile_dim1/tile_dim2 unused
  g1[5] = 16384;                // tensor_dim0_stride[31:0]
  g1[6] = 0;
  g1[7] = 0;
  tv4i z4 = {0, 0, 0, 0};
#if __clang_major__ >= 23
  tv8i z8 = {0, 0, 0, 0, 0, 0, 0, 0};
  __builtin_amdgcn_tensor_load_to_lds(g0, g1, z4, z4, z8, 0);
#else
  __builtin_amdgcn_tensor_load_to_lds(g0, g1, z4, z4, 0);
#endif
}
#endif

__device__ __forceinline__ v16h ldfrag2(const _Float16* p0, const _Float16* p1) {
  v8h a = *(const v8h*)p0;
  v8h b = *(const v8h*)p1;
  return __builtin_shufflevector(a, b, 0,1,2,3,4,5,6,7,8,9,10,11,12,13,14,15);
}

__device__ __forceinline__ v16h ldcolB(const _Float16* p) {
  v16h r;
#pragma unroll
  for (int s = 0; s < 8; ++s) {
    r[s] = p[s * Z_PITCH];
    r[8 + s] = p[(16 + s) * Z_PITCH];
  }
  return r;
}

// ---------------------------------------------------------------------------
// Kernel 1: embed_weight f32[K,C] -> f16 (row-major) and wnorm[k] = ||w_k||^2
// ---------------------------------------------------------------------------
__global__ __launch_bounds__(256) void vq_prep_embed(
    const float* __restrict__ w, _Float16* __restrict__ wf16,
    float* __restrict__ wnorm)
{
  const int lane = threadIdx.x & 31;
  const int code = blockIdx.x * 8 + (threadIdx.x >> 5);
  float s = 0.f;
#pragma unroll
  for (int j = 0; j < 8; ++j) {
    int c = lane + 32 * j;
    float v = w[code * CCH + c];
    s += v * v;
    wf16[code * CCH + c] = (_Float16)v;
  }
#pragma unroll
  for (int off = 16; off >= 1; off >>= 1)
    s += __shfl_xor(s, off, 32);
  if (lane == 0) wnorm[code] = s;
}

// ---------------------------------------------------------------------------
// Kernel 2: z[B,C,D,H,W] f32 -> zt f16 in BLOCK-TRANSPOSED layout:
//   zt[(n/128)][c][n%128]  (each 128-row block is a contiguous 64KB slab,
//   already channel-major => GEMM stages it with pure vector copies)
// ---------------------------------------------------------------------------
__global__ __launch_bounds__(256) void vq_prep_z(
    const float* __restrict__ z, _Float16* __restrict__ zt)
{
  const int b    = blockIdx.x >> 8;            // 256 blocks per batch
  const int dhw0 = (blockIdx.x & 255) * 32;
  const int dhw  = dhw0 + (threadIdx.x & 31);
  const int g    = threadIdx.x >> 5;           // 0..7
  const int n    = b * DHW + dhw;
  const size_t blkbase = (size_t)(n >> 7) * (CCH * TILE_R);
  const int r = n & 127;
#pragma unroll
  for (int j = 0; j < 32; ++j) {
    int c = g + 8 * j;
    float v = z[(b * CCH + c) * DHW + dhw];    // coalesced per 32-lane group
    zt[blkbase + (size_t)c * TILE_R + r] = (_Float16)v;
  }
}

// ---------------------------------------------------------------------------
// Kernel 3: WMMA GEMM + fused argmin.
//   A (M=codes) from row-major codebook tile, B (N=z-rows) from pre-transposed
//   z slab. W tiles double-buffered: TDM (one descriptor per 64KB tile) when
//   available, else per-thread async loads. dist[m][n] = wnorm[m] - 2*S[m][n];
//   running per-z-row argmin in registers, final 8-wave combine through LDS.
// ---------------------------------------------------------------------------
__global__ __launch_bounds__(256) void vq_gemm_argmin(
    const _Float16* __restrict__ zt, const _Float16* __restrict__ wf16,
    const float* __restrict__ wnorm, int* __restrict__ out_idx)
{
  extern __shared__ __align__(16) char smem_raw[];
  _Float16* shZ = (_Float16*)smem_raw;               // [CCH][Z_PITCH]
  _Float16* shW0 = shZ + CCH * Z_PITCH;              // [TILE_C][W_PITCH] x2
  _Float16* shW1 = shW0 + TILE_C * W_PITCH;

  const int tid  = threadIdx.x;
  const int wave = tid >> 5;
  const int lane = tid & 31;
  const int half = lane >> 4;                        // 0: codes v, 1: codes v+8
  const int l15  = lane & 15;
  const int lo   = half ? 8 : 0;
  const int rb   = blockIdx.x * TILE_R;              // this block's z rows

  // ---- stage Z slab (64KB contiguous) once: 16 x b128 per thread ----
  {
    const _Float16* gz = zt + (size_t)blockIdx.x * (CCH * TILE_R);
#pragma unroll
    for (int it = 0; it < 16; ++it) {
      int i = tid + it * 256;                        // 4096 x uint4
      int r = i >> 4, seg = i & 15;                  // 16 x uint4 per 128h row
      cp16(gz + i * 8, shZ + r * Z_PITCH + seg * 8);
    }
  }
  // ---- stage W tile 0 ----
  {
    const _Float16* gw = wf16;
#pragma unroll
    for (int it = 0; it < 16; ++it) {
      int i = tid + it * 256;                        // 4096 x uint4
      int r = i >> 5, seg = i & 31;                  // 32 x uint4 per 256h row
      cp16(gw + i * 8, shW0 + r * W_PITCH + seg * 8);
    }
  }

  float best[8];
  int   bidx[8];
#pragma unroll
  for (int nc = 0; nc < 8; ++nc) { best[nc] = 3.0e38f; bidx[nc] = 0; }

  for (int t = 0; t < NPASS; ++t) {
#if USE_ASYNC
    __builtin_amdgcn_s_wait_asynccnt(0);             // prologue tiles landed
#endif
#if USE_TDM
    __builtin_amdgcn_s_wait_tensorcnt(0);            // TDM tile t landed
#endif
    __syncthreads();                                 // publish; prior pass done
    _Float16* shW = (t & 1) ? shW1 : shW0;

    // issue/stage next W tile into the other buffer (overlaps WMMA below)
    if (t + 1 < NPASS) {
      const _Float16* gw = wf16 + (size_t)(t + 1) * (TILE_C * CCH);
      _Float16* dst = (t & 1) ? shW0 : shW1;
#if USE_TDM
      if (wave == 0) tdm_load_tile(gw, dst);         // one D# moves 64KB
#else
#pragma unroll
      for (int it = 0; it < 16; ++it) {
        int i = tid + it * 256;
        int r = i >> 5, seg = i & 31;
        cp16(gw + i * 8, dst + r * W_PITCH + seg * 8);
      }
#endif
    }

    const int mb = t * TILE_C + wave * 16;           // wave's 16 codes

    // ---- A fragments: this wave's 16 code rows (ISA 16-bit A 16x32 layout)
    v16h afrag[8];
#pragma unroll
    for (int k8 = 0; k8 < 8; ++k8) {
      const _Float16* ap = shW + (wave * 16 + l15) * W_PITCH + k8 * 32 + lo;
      afrag[k8] = ldfrag2(ap, ap + 16);
    }
    float wn[8];
#pragma unroll
    for (int v = 0; v < 8; ++v) wn[v] = wnorm[mb + v + 8 * half];

#pragma unroll
    for (int nc = 0; nc < 8; ++nc) {
      v8f acc = {0.f, 0.f, 0.f, 0.f, 0.f, 0.f, 0.f, 0.f};
      // pipelined B fragments (dense 32x16 B layout: lane%16 = N, elements = K in table order)
      const _Float16* bp0 = shZ + lo * Z_PITCH + nc * 16 + l15;
      v16h bf = ldcolB(bp0);
#pragma unroll
      for (int k8 = 0; k8 < 8; ++k8) {
        v16h bnext;
        if (k8 < 7) {
          const _Float16* bp = shZ + ((k8 + 1) * 32 + lo) * Z_PITCH + nc * 16 + l15;
          bnext = ldcolB(bp);
        }
        acc = __builtin_amdgcn_wmma_f32_16x16x32_f16(
            false, afrag[k8], false, bf, (short)0, acc, false, false);
        if (k8 < 7) bf = bnext;
      }
      const int cbase = mb + 8 * half;               // codes this lane holds
#pragma unroll
      for (int v = 0; v < 8; ++v) {
        float d = fmaf(-2.0f, acc[v], wn[v]);
        if (d < best[nc]) { best[nc] = d; bidx[nc] = cbase + v; }
      }
    }
  }

  // ---- merge the two 16-lane halves (codes v vs v+8) per z-row ----
#pragma unroll
  for (int nc = 0; nc < 8; ++nc) {
    float od = __shfl_xor(best[nc], 16, 32);
    int   oi = __shfl_xor(bidx[nc], 16, 32);
    if (od < best[nc] || (od == best[nc] && oi < bidx[nc])) {
      best[nc] = od; bidx[nc] = oi;
    }
  }
  __syncthreads();                                   // smem reuse for reduce
  float* sbest = (float*)smem_raw;                   // [8 waves][128 rows]
  int*   sidx  = (int*)(smem_raw + 8 * TILE_R * 4);
  if (half == 0) {
#pragma unroll
    for (int nc = 0; nc < 8; ++nc) {
      sbest[wave * TILE_R + nc * 16 + l15] = best[nc];
      sidx [wave * TILE_R + nc * 16 + l15] = bidx[nc];
    }
  }
  __syncthreads();
  if (tid < TILE_R) {                                // combine 8 waves' codes
    float bb = sbest[tid]; int bi = sidx[tid];
#pragma unroll
    for (int w = 1; w < 8; ++w) {
      float ob = sbest[w * TILE_R + tid]; int oi = sidx[w * TILE_R + tid];
      if (ob < bb || (ob == bb && oi < bi)) { bb = ob; bi = oi; }
    }
    out_idx[rb + tid] = bi;
  }
}

// ---------------------------------------------------------------------------
// Kernel 4: gather z_q (fp32 weights), write z_q_st = z + (z_q - z),
// float indices, and per-block loss partial sums.
// ---------------------------------------------------------------------------
__global__ __launch_bounds__(256) void vq_gather_loss(
    const float* __restrict__ z, const float* __restrict__ w,
    const int* __restrict__ idx, float* __restrict__ out,
    float* __restrict__ out_ind, float* __restrict__ partials)
{
  __shared__ float sred[256];
  const int tid  = threadIdx.x;
  const int b    = blockIdx.x >> 8;
  const int dhw0 = (blockIdx.x & 255) * 32;
  const int dhw  = dhw0 + (tid & 31);
  const int n    = b * DHW + dhw;
  const int c0   = tid >> 5;
  const int myidx = idx[n];
  if (tid < 32) out_ind[n] = (float)myidx;

  float acc = 0.f;
#pragma unroll
  for (int j = 0; j < 32; ++j) {
    int c = c0 * 32 + j;
    float zq   = w[myidx * CCH + c];
    float zv   = z[(b * CCH + c) * DHW + dhw];
    float diff = zq - zv;
    out[(b * CCH + c) * DHW + dhw] = zv + diff;   // straight-through forward
    acc += diff * diff;
  }
  sred[tid] = acc;
  __syncthreads();
#pragma unroll
  for (int s = 128; s >= 1; s >>= 1) {
    if (tid < s) sred[tid] += sred[tid + s];
    __syncthreads();
  }
  if (tid == 0) partials[blockIdx.x] = sred[0];
}

__global__ __launch_bounds__(256) void vq_finalize(
    const float* __restrict__ partials, float* __restrict__ loss_out)
{
  __shared__ float sred[256];
  const int tid = threadIdx.x;
  float s = 0.f;
  for (int i = tid; i < 2048; i += 256) s += partials[i];
  sred[tid] = s;
  __syncthreads();
#pragma unroll
  for (int k = 128; k >= 1; k >>= 1) {
    if (tid < k) sred[tid] += sred[tid + k];
    __syncthreads();
  }
  // loss = mean((zq-z)^2) + BETA*mean((zq-z)^2), BETA = 0.25
  if (tid == 0) loss_out[0] = 1.25f * sred[0] / (float)NELEM;
}

// ---------------------------------------------------------------------------
extern "C" void kernel_launch(void* const* d_in, const int* in_sizes, int n_in,
                              void* d_out, int out_size, void* d_ws, size_t ws_size,
                              hipStream_t stream) {
  (void)in_sizes; (void)n_in; (void)out_size; (void)ws_size;
  const float* z    = (const float*)d_in[0];   // [4,256,16,32,32]
  const float* wemb = (const float*)d_in[1];   // [4096,256]

  float* out      = (float*)d_out;             // z_q_st: 16777216 floats
  float* out_loss = out + NELEM;               // 1 float
  float* out_ind  = out + NELEM + 1;           // 65536 floats (indices)

  char* ws = (char*)d_ws;                      // ~36 MB of scratch
  _Float16* ztb    = (_Float16*)(ws);                         // 32 MB
  _Float16* wf16   = (_Float16*)(ws + 33554432);              // 2 MB
  float*    wnorm  = (float*)   (ws + 35651584);              // 16 KB
  int*      idxbuf = (int*)     (ws + 35667968);              // 256 KB
  float*    parts  = (float*)   (ws + 35930112);              // 8 KB

  vq_prep_embed <<<dim3(KCODES / 8), dim3(256), 0, stream>>>(wemb, wf16, wnorm);
  vq_prep_z     <<<dim3(NROWS / 32), dim3(256), 0, stream>>>(z, ztb);
  vq_gemm_argmin<<<dim3(NROWS / TILE_R), dim3(256), SMEM_BYTES, stream>>>(
      ztb, wf16, wnorm, idxbuf);
  vq_gather_loss<<<dim3(NROWS / 32), dim3(256), 0, stream>>>(
      z, wemb, idxbuf, out, out_ind, parts);
  vq_finalize   <<<dim3(1), dim3(256), 0, stream>>>(parts, out_loss);
}
